// MCLAMDA_61469571940892
// MI455X (gfx1250) — compile-verified, matrix operand order checked by reading, not run on hardware
//
#include <hip/hip_runtime.h>
#include <math.h>

typedef __attribute__((ext_vector_type(16))) __bf16 v16bf;
typedef __attribute__((ext_vector_type(8)))  __bf16 v8bf;
typedef __attribute__((ext_vector_type(8)))  float  v8f;

#define D_DIM  256
#define NDIS   8192
#define NMIR   16384
#define NTOT   24576
#define BPAIR  65536
#define KNN    50
#define NEG_INF (-3.402823466e38f)

struct bf4 { __bf16 x, y, z, w; };

// ---------------------------------------------------------------------------
// f32 -> bf16 conversion (emb slab lives in L2 for the O(N^2) WMMA passes)
// ---------------------------------------------------------------------------
__global__ __launch_bounds__(256) void cvt_bf16_kernel(const float* __restrict__ in,
                                                       __bf16* __restrict__ out) {
  size_t i = ((size_t)blockIdx.x * 256 + threadIdx.x) * 4;
  float4 f = *(const float4*)(in + i);
  bf4 o;
  o.x = (__bf16)f.x; o.y = (__bf16)f.y; o.z = (__bf16)f.z; o.w = (__bf16)f.w;
  *(bf4*)(out + i) = o;
}

// ---------------------------------------------------------------------------
// WMMA fragment load: 16-bit A-matrix 16x32 layout (lanes 0-15: K {0-7,16-23},
// lanes 16-31: K {8-15,24-31}).  B of F*F^T is F^T, whose B-layout equals the
// A-layout of F, so both fragments load with the same addressing.
// ---------------------------------------------------------------------------
__device__ __forceinline__ v16bf load_frag(const __bf16* slab, int row, int kc, int g) {
  const __bf16* p = slab + (size_t)row * D_DIM + kc * 32 + g * 8;
  union { v16bf v; struct { v8bf lo; v8bf hi; } s; } u;
  u.s.lo = *(const v8bf*)(p);
  u.s.hi = *(const v8bf*)(p + 16);
  return u.v;
}

// ---------------------------------------------------------------------------
// Fused  sim = F*F^T  (bf16 WMMA, f32 acc)  +  streaming top-51  +  50-NN mean.
// One workgroup (8 wave32) per 16 rows; each wave owns column tiles ct = wv+8k.
// Software-pipelined: ping-pong fragment buffers so each tile's 16 b128 loads
// are issued one full phase (WMMA chain + top-k insert) before they are
// consumed; the asm "memory" barrier in the insert phase pins load issue.
// ---------------------------------------------------------------------------
__global__ __launch_bounds__(256) void knn_update_kernel(
    const __bf16* __restrict__ fb,   // slab bf16, Nf x 256
    const float*  __restrict__ ff,   // slab f32,  Nf x 256
    float* __restrict__ upd,         // Nf x 256
    int Nf)
{
  __shared__ float stage[8][16][16];
  __shared__ float listV[8][16][51];
  __shared__ int   listI[8][16][51];
  __shared__ int   nbrIdx[16][KNN];

  const int tid  = threadIdx.x;
  const int lane = tid & 31;
  const int wv   = tid >> 5;
  const int g    = lane >> 4;
  const int lr   = lane & 15;
  const int rowBase = blockIdx.x * 16;

  if (lane < 16) {
    for (int j = 0; j < 51; ++j) { listV[wv][lane][j] = NEG_INF; listI[wv][lane][j] = 0; }
  }
  float minv = NEG_INF;
  int   minpos = 0;

  // A fragments for this workgroup's 16 rows, all 256 K
  v16bf af[8];
  #pragma unroll
  for (int kc = 0; kc < 8; ++kc) af[kc] = load_frag(fb, rowBase + lr, kc, g);

  // WMMA chain + streaming top-51 insert for one 16x16 sim tile
  auto process_tile = [&](const v16bf (&bfr)[8], int cbase) {
    v8f acc0 = {0.f, 0.f, 0.f, 0.f, 0.f, 0.f, 0.f, 0.f};
    v8f acc1 = {0.f, 0.f, 0.f, 0.f, 0.f, 0.f, 0.f, 0.f};
    #pragma unroll
    for (int kc = 0; kc < 4; ++kc) {
      acc0 = __builtin_amdgcn_wmma_f32_16x16x32_bf16(false, af[2 * kc],     false, bfr[2 * kc],
                                                     (short)0, acc0, false, false);
      acc1 = __builtin_amdgcn_wmma_f32_16x16x32_bf16(false, af[2 * kc + 1], false, bfr[2 * kc + 1],
                                                     (short)0, acc1, false, false);
    }
    v8f acc = acc0 + acc1;

    // C layout: VGPR v, lanes 0-15 -> M=v, lanes 16-31 -> M=v+8; N = lane%16
    #pragma unroll
    for (int v = 0; v < 8; ++v) stage[wv][v + 8 * g][lr] = acc[v];
    asm volatile("s_wait_dscnt 0x0" ::: "memory");  // wave-local LDS RAW + sched fence

    if (lane < 16) {
      const int r = lane;
      for (int c = 0; c < 16; ++c) {
        float val = stage[wv][r][c];
        if (val > minv) {
          listV[wv][r][minpos] = val;
          listI[wv][r][minpos] = cbase + c;
          float nm = listV[wv][r][0]; int np = 0;
          for (int j = 1; j < 51; ++j) {
            float x = listV[wv][r][j];
            if (x < nm) { nm = x; np = j; }
          }
          minv = nm; minpos = np;
        }
      }
    }
  };

  const int nTiles = Nf >> 4;            // 512 or 1024
  const int tilesPerWave = nTiles >> 3;  // 64 or 128 (even)

  v16bf bufA[8], bufB[8];
  #pragma unroll
  for (int kc = 0; kc < 8; ++kc) bufA[kc] = load_frag(fb, (wv << 4) + lr, kc, g);

  for (int i = 0; i < tilesPerWave; i += 2) {
    const int ct0 = wv + i * 8;
    const int ct1 = ct0 + 8;   // always < nTiles (tilesPerWave even)
    const int ct2 = ct0 + 16;

    if (ct2 < nTiles)
      __builtin_prefetch(fb + (size_t)((ct2 << 4) + lr) * D_DIM + g * 128, 0, 0);

    #pragma unroll
    for (int kc = 0; kc < 8; ++kc) bufB[kc] = load_frag(fb, (ct1 << 4) + lr, kc, g);

    process_tile(bufA, ct0 << 4);

    if (ct2 < nTiles) {
      #pragma unroll
      for (int kc = 0; kc < 8; ++kc) bufA[kc] = load_frag(fb, (ct2 << 4) + lr, kc, g);
    }

    process_tile(bufB, ct1 << 4);
  }
  __syncthreads();

  // merge 8 per-wave lists (408 candidates) per row; wave wv -> rows 2wv, 2wv+1
  for (int rsel = 0; rsel < 2; ++rsel) {
    const int rr = wv * 2 + rsel;
    float mv[13]; int mi[13];
    #pragma unroll
    for (int e = 0; e < 13; ++e) {
      int t = lane + 32 * e;
      if (t < 408) {
        int w2 = t / 51; int j = t - w2 * 51;
        mv[e] = listV[w2][rr][j]; mi[e] = listI[w2][rr][j];
      } else { mv[e] = NEG_INF; mi[e] = 0; }
    }
    for (int t = 0; t < 51; ++t) {
      float bv = NEG_INF; int bidx = 0; int bkey = lane << 4;
      #pragma unroll
      for (int e = 0; e < 13; ++e)
        if (mv[e] > bv) { bv = mv[e]; bidx = mi[e]; bkey = (lane << 4) | e; }
      #pragma unroll
      for (int off = 16; off >= 1; off >>= 1) {
        float ov = __shfl_xor(bv, off);
        int   oi = __shfl_xor(bidx, off);
        int   ok = __shfl_xor(bkey, off);
        if (ov > bv || (ov == bv && ok < bkey)) { bv = ov; bidx = oi; bkey = ok; }
      }
      if ((bkey >> 4) == lane) {
        int slot = bkey & 15;
        #pragma unroll
        for (int e = 0; e < 13; ++e) if (e == slot) mv[e] = NEG_INF;
      }
      if (t > 0 && lane == 0) nbrIdx[rr][t - 1] = bidx;  // drop rank 0 (self)
    }
  }
  __syncthreads();

  // mean of 50 neighbor feature rows (f32 source); thread = dim (D==256)
  const int d = tid;
  for (int r = 0; r < 16; ++r) {
    float s = 0.f;
    for (int j = 0; j < KNN; ++j)
      s += ff[(size_t)nbrIdx[r][j] * D_DIM + d];
    upd[(size_t)(rowBase + r) * D_DIM + d] = s * (1.0f / KNN);
  }
}

// ---------------------------------------------------------------------------
// column mean of upd (one block, thread = dim)
// ---------------------------------------------------------------------------
__global__ __launch_bounds__(256) void center_kernel(const float* __restrict__ upd,
                                                     int n, float* __restrict__ center) {
  const int d = threadIdx.x;
  float s = 0.f;
  for (int r = 0; r < n; ++r) s += upd[(size_t)r * D_DIM + d];
  center[d] = s / (float)n;
}

// ---------------------------------------------------------------------------
// per-row softplus(cos(row, other) - cos(row, own)); wave per row; one
// partial per block (deterministic, no float atomics)
// ---------------------------------------------------------------------------
__global__ __launch_bounds__(256) void knn_loss_kernel(
    const float* __restrict__ feat,
    const float* __restrict__ cOwn, const float* __restrict__ cOth,
    float* __restrict__ partial)
{
  __shared__ float sh[8];
  const int lane = threadIdx.x & 31, wv = threadIdx.x >> 5;
  const int row = blockIdx.x * 8 + wv;
  const float4* x4 = (const float4*)(feat + (size_t)row * D_DIM);
  const float4* o4 = (const float4*)cOwn;
  const float4* t4 = (const float4*)cOth;
  float xx = 0, xo = 0, xt = 0, oo = 0, tt = 0;
  #pragma unroll
  for (int t = 0; t < 2; ++t) {
    float4 xv = x4[lane + 32 * t], ov = o4[lane + 32 * t], tv = t4[lane + 32 * t];
    xx += xv.x * xv.x + xv.y * xv.y + xv.z * xv.z + xv.w * xv.w;
    xo += xv.x * ov.x + xv.y * ov.y + xv.z * ov.z + xv.w * ov.w;
    xt += xv.x * tv.x + xv.y * tv.y + xv.z * tv.z + xv.w * tv.w;
    oo += ov.x * ov.x + ov.y * ov.y + ov.z * ov.z + ov.w * ov.w;
    tt += tv.x * tv.x + tv.y * tv.y + tv.z * tv.z + tv.w * tv.w;
  }
  #pragma unroll
  for (int off = 16; off >= 1; off >>= 1) {
    xx += __shfl_xor(xx, off); xo += __shfl_xor(xo, off); xt += __shfl_xor(xt, off);
    oo += __shfl_xor(oo, off); tt += __shfl_xor(tt, off);
  }
  if (lane == 0) {
    const float eps = 1e-8f;
    float nx = fmaxf(sqrtf(xx), eps), no = fmaxf(sqrtf(oo), eps), nt = fmaxf(sqrtf(tt), eps);
    float a = xt / (nx * nt) - xo / (nx * no);
    sh[wv] = fmaxf(a, 0.f) + log1pf(expf(-fabsf(a)));
  }
  __syncthreads();
  if (threadIdx.x == 0) {
    float s = 0.f;
    #pragma unroll
    for (int i = 0; i < 8; ++i) s += sh[i];
    partial[blockIdx.x] = s;
  }
}

// ---------------------------------------------------------------------------
// scores (sigmoid of h_d . h_m) fused with per-pair contrastive term
// ---------------------------------------------------------------------------
__global__ __launch_bounds__(256) void pair_kernel(
    const float* __restrict__ h, const float* __restrict__ src,
    const float* __restrict__ dst, const int* __restrict__ dis,
    const int* __restrict__ mir, float* __restrict__ scores,
    float* __restrict__ partial)
{
  __shared__ float sh[8];
  const int lane = threadIdx.x & 31, wv = threadIdx.x >> 5;
  const int b = blockIdx.x * 8 + wv;
  const int di = dis[b], mj = mir[b];
  const float4* hd4 = (const float4*)(h + (size_t)di * D_DIM);
  const float4* hm4 = (const float4*)(h + (size_t)mj * D_DIM);
  const float4* sr4 = (const float4*)(src + (size_t)b * D_DIM);
  const float4* dr4 = (const float4*)(dst + (size_t)b * D_DIM);
  float dm = 0, hh = 0, ss = 0, dd = 0, hs = 0, hdd = 0;
  #pragma unroll
  for (int t = 0; t < 2; ++t) {
    float4 a = hd4[lane + 32 * t], m = hm4[lane + 32 * t];
    float4 s = sr4[lane + 32 * t], dv = dr4[lane + 32 * t];
    dm  += a.x * m.x + a.y * m.y + a.z * m.z + a.w * m.w;
    hh  += a.x * a.x + a.y * a.y + a.z * a.z + a.w * a.w;
    ss  += s.x * s.x + s.y * s.y + s.z * s.z + s.w * s.w;
    dd  += dv.x * dv.x + dv.y * dv.y + dv.z * dv.z + dv.w * dv.w;
    hs  += a.x * s.x + a.y * s.y + a.z * s.z + a.w * s.w;
    hdd += a.x * dv.x + a.y * dv.y + a.z * dv.z + a.w * dv.w;
  }
  #pragma unroll
  for (int off = 16; off >= 1; off >>= 1) {
    dm += __shfl_xor(dm, off); hh += __shfl_xor(hh, off); ss += __shfl_xor(ss, off);
    dd += __shfl_xor(dd, off); hs += __shfl_xor(hs, off); hdd += __shfl_xor(hdd, off);
  }
  if (lane == 0) {
    scores[b] = 1.f / (1.f + expf(-dm));
    const float eps = 1e-8f;
    float nh = fmaxf(sqrtf(hh), eps), ns = fmaxf(sqrtf(ss), eps), nd = fmaxf(sqrtf(dd), eps);
    float x = hdd / (nh * nd) - hs / (nh * ns);   // neg_sim - pos_sim
    sh[wv] = fmaxf(x, 0.f) + log1pf(expf(-fabsf(x)));
  }
  __syncthreads();
  if (threadIdx.x == 0) {
    float s = 0.f;
    #pragma unroll
    for (int i = 0; i < 8; ++i) s += sh[i];
    partial[blockIdx.x] = s;
  }
}

// ---------------------------------------------------------------------------
// deterministic reduction of all partials -> the two scalar losses
// ---------------------------------------------------------------------------
__global__ __launch_bounds__(256) void finalize_kernel(
    const float* __restrict__ pC, int nC, const float* __restrict__ pA, int nA,
    const float* __restrict__ pB, int nB, float* __restrict__ out)
{
  __shared__ float sh[256];
  const float* ps[3] = { pC, pA, pB };
  int ns[3] = { nC, nA, nB };
  float sums[3];
  for (int q = 0; q < 3; ++q) {
    float s = 0.f;
    for (int i = threadIdx.x; i < ns[q]; i += 256) s += ps[q][i];
    sh[threadIdx.x] = s;
    __syncthreads();
    for (int off = 128; off >= 1; off >>= 1) {
      if (threadIdx.x < off) sh[threadIdx.x] += sh[threadIdx.x + off];
      __syncthreads();
    }
    sums[q] = sh[0];
    __syncthreads();
  }
  if (threadIdx.x == 0) {
    out[BPAIR]     = sums[0] / (float)BPAIR;
    out[BPAIR + 1] = 0.5f * (sums[1] / (float)NDIS + sums[2] / (float)NMIR);
  }
}

// ---------------------------------------------------------------------------
extern "C" void kernel_launch(void* const* d_in, const int* in_sizes, int n_in,
                              void* d_out, int out_size, void* d_ws, size_t ws_size,
                              hipStream_t stream) {
  (void)in_sizes; (void)n_in; (void)out_size; (void)ws_size;
  const float* h   = (const float*)d_in[0];
  const float* emb = (const float*)d_in[1];
  const float* src = (const float*)d_in[2];
  const float* dst = (const float*)d_in[3];
  const int*   dis = (const int*)d_in[4];
  const int*   mir = (const int*)d_in[5];
  float* out = (float*)d_out;

  char* ws = (char*)d_ws;
  // workspace layout (256B aligned)
  constexpr size_t OFF_EMBBF = 0;                                    // 24576*256*2
  constexpr size_t OFF_UPDD  = OFF_EMBBF + (size_t)NTOT * D_DIM * 2; // 8192*256*4
  constexpr size_t OFF_UPDM  = OFF_UPDD + (size_t)NDIS * D_DIM * 4;  // 16384*256*4
  constexpr size_t OFF_CA    = OFF_UPDM + (size_t)NMIR * D_DIM * 4;
  constexpr size_t OFF_CB    = OFF_CA + 1024;
  constexpr size_t OFF_PC    = OFF_CB + 1024;                        // 8192 floats
  constexpr size_t OFF_PA    = OFF_PC + (BPAIR / 8) * 4;             // 1024 floats
  constexpr size_t OFF_PB    = OFF_PA + (NDIS / 8) * 4;              // 2048 floats

  __bf16* embBf  = (__bf16*)(ws + OFF_EMBBF);
  float*  updD   = (float*)(ws + OFF_UPDD);
  float*  updM   = (float*)(ws + OFF_UPDM);
  float*  cA     = (float*)(ws + OFF_CA);
  float*  cB     = (float*)(ws + OFF_CB);
  float*  pC     = (float*)(ws + OFF_PC);
  float*  pA     = (float*)(ws + OFF_PA);
  float*  pB     = (float*)(ws + OFF_PB);

  cvt_bf16_kernel<<<(NTOT * D_DIM) / (256 * 4), 256, 0, stream>>>(emb, embBf);

  knn_update_kernel<<<NDIS / 16, 256, 0, stream>>>(embBf, emb, updD, NDIS);
  knn_update_kernel<<<NMIR / 16, 256, 0, stream>>>(
      embBf + (size_t)NDIS * D_DIM, emb + (size_t)NDIS * D_DIM, updM, NMIR);

  center_kernel<<<1, 256, 0, stream>>>(updD, NDIS, cA);
  center_kernel<<<1, 256, 0, stream>>>(updM, NMIR, cB);

  knn_loss_kernel<<<NDIS / 8, 256, 0, stream>>>(updD, cA, cB, pA);
  knn_loss_kernel<<<NMIR / 8, 256, 0, stream>>>(updM, cB, cA, pB);

  pair_kernel<<<BPAIR / 8, 256, 0, stream>>>(h, src, dst, dis, mir, out, pC);

  finalize_kernel<<<1, 256, 0, stream>>>(pC, BPAIR / 8, pA, NDIS / 8, pB, NMIR / 8, out);
}
